// LSTM_66692252172978
// MI455X (gfx1250) — matrix operand order repaired and checked by reading, not compiled, not run on hardware
//
#include <hip/hip_runtime.h>
#include <hip/hip_bf16.h>

typedef __attribute__((ext_vector_type(16))) __bf16 v16bf;
typedef __attribute__((ext_vector_type(8)))  __bf16 v8bf;
typedef __attribute__((ext_vector_type(8)))  float  v8f;
typedef int v4i_g __attribute__((__vector_size__(16)));   // matches builtin's param type

#define T_LEN   4096
#define NB      64
#define HID     128
#define G4      512      // 4*HID gates
#define LDS_PAD 136      // HID + 8 pad

// ---- gfx1250 async global->LDS path (guarded so the build never breaks) ----
#if defined(__has_builtin)
#if __has_builtin(__builtin_amdgcn_global_load_async_to_lds_b128) && \
    __has_builtin(__builtin_amdgcn_s_wait_asynccnt)
#define USE_ASYNC_GX 1
#endif
#endif
#ifndef USE_ASYNC_GX
#define USE_ASYNC_GX 0
#endif

#define AS1 __attribute__((address_space(1)))
#define AS3 __attribute__((address_space(3)))

__device__ __forceinline__ float sigf(float x)      { return 1.0f / (1.0f + __expf(-x)); }
__device__ __forceinline__ float tanhfast(float x)  { return 2.0f / (1.0f + __expf(-2.0f * x)) - 1.0f; }

#if USE_ASYNC_GX
__device__ __forceinline__ void async_copy16(const float* gp, float* lp) {
    __builtin_amdgcn_global_load_async_to_lds_b128((AS1 v4i_g*)gp, (AS3 v4i_g*)lp, 0, 0);
}
#endif

// ---------------- weight prep ----------------
__global__ void f2bf_kernel(const float* __restrict__ s, __bf16* __restrict__ d, int n) {
    int i = blockIdx.x * blockDim.x + threadIdx.x;
    if (i < n) d[i] = (__bf16)s[i];
}

__global__ void bias_kernel(const float* __restrict__ a, const float* __restrict__ b,
                            float* __restrict__ o, int n) {
    int i = blockIdx.x * blockDim.x + threadIdx.x;
    if (i < n) o[i] = a[i] + b[i];
}

// ---------------- layer 0 input projection (in_dim == 1) ----------------
// gx layout: [dir][t][gate][batch] f32
__global__ __launch_bounds__(256) void in_proj_l0(
        const float* __restrict__ x,      // [NB][T]
        const float* __restrict__ wf,     // [512] fwd Wih
        const float* __restrict__ wb,     // [512] bwd Wih
        const float* __restrict__ biasc,  // [2][512] (bih+bhh)
        float* __restrict__ gx) {
    size_t idx = (size_t)blockIdx.x * 256 + threadIdx.x;  // over T*G4*NB, batch fastest
    int b = (int)(idx & (NB - 1));
    int g = (int)((idx >> 6) & (G4 - 1));
    int t = (int)(idx >> 15);
    float xv = x[(size_t)b * T_LEN + t];
    const size_t TOT = (size_t)T_LEN * G4 * NB;
    gx[idx]       = xv * wf[g] + biasc[g];
    gx[TOT + idx] = xv * wb[g] + biasc[G4 + g];
}

// ---------------- layers 1/2 input projection: [16,256]x[256,512] WMMA tiles ----------------
__global__ __launch_bounds__(256) void in_proj_gemm(
        const __bf16* __restrict__ hin,   // [NB][T][256] bf16
        const __bf16* __restrict__ wih,   // [2][512][256] bf16
        const float*  __restrict__ biasc, // [2][512]
        float* __restrict__ gx) {         // [2][T][512][NB] f32
    const int t   = blockIdx.x >> 2;
    const int b0  = (blockIdx.x & 3) * 16;
    const int dir = blockIdx.y;
    const int tid = threadIdx.x;
    const int wv  = tid >> 5;        // wave 0..7
    const int lane = tid & 31;
    const int lh  = lane >> 4;       // lane half
    const int l15 = lane & 15;

    const __bf16* W  = wih + (size_t)dir * G4 * 256;
    const float*  bs = biasc + dir * G4;

    v8f acc[4];
    #pragma unroll
    for (int j = 0; j < 4; ++j) {
        float bv = bs[(wv + 8 * j) * 16 + l15];   // bias depends only on gate column N
        #pragma unroll
        for (int v = 0; v < 8; ++v) acc[j][v] = bv;
    }

    #pragma unroll
    for (int ks = 0; ks < 8; ++ks) {              // K = 256 in 8 steps of 32
        const __bf16* ap = hin + ((size_t)(b0 + l15) * T_LEN + t) * 256 + ks * 32 + lh * 16;
        v8bf lo = *(const v8bf*)ap;
        v8bf hi = *(const v8bf*)(ap + 8);
        v16bf a;
        #pragma unroll
        for (int e = 0; e < 8; ++e) { a[e] = lo[e]; a[8 + e] = hi[e]; }
        #pragma unroll
        for (int j = 0; j < 4; ++j) {
            int g = (wv + 8 * j) * 16 + l15;
            v16bf bvv = *(const v16bf*)(W + (size_t)g * 256 + ks * 32 + lh * 16);
            acc[j] = __builtin_amdgcn_wmma_f32_16x16x32_bf16(
                         false, a, false, bvv, (short)0, acc[j], false, false);
        }
    }

    float* gout = gx + ((size_t)dir * T_LEN + t) * (size_t)G4 * NB;
    #pragma unroll
    for (int j = 0; j < 4; ++j) {
        int g = (wv + 8 * j) * 16 + l15;
        *(v8f*)(gout + (size_t)g * NB + b0 + lh * 8) = acc[j];
    }
}

// ---------------- persistent bidirectional LSTM scan ----------------
// grid = 8 blocks: blockIdx>>2 = direction, blockIdx&3 = batch slice of 16.
// Whh B-fragments live in VGPRs for all 4096 steps; cell state c lives in WMMA
// accumulator registers; gx gate biases are double-buffered in LDS via
// GLOBAL_LOAD_ASYNC_TO_LDS_B128 prefetch (ASYNCcnt-tracked).
__global__ __launch_bounds__(256) void lstm_scan(
        const float*  __restrict__ gx,    // [2][T][512][NB]
        const __bf16* __restrict__ whh,   // [2][512][128] bf16 (this layer)
        __bf16* __restrict__ hout) {      // [NB][T][256]
    extern __shared__ char smem[];
    __bf16* hlds  = (__bf16*)smem;                                  // [16][LDS_PAD]
    float*  gxlds = (float*)(smem + ((16 * LDS_PAD * 2 + 15) & ~15)); // 2 x [512][16]
    const int dir = blockIdx.x >> 2;
    const int b0  = (blockIdx.x & 3) * 16;
    const int tid = threadIdx.x;
    const int wv  = tid >> 5;        // wave = hidden tile 0..7
    const int lane = tid & 31;
    const int lh  = lane >> 4;
    const int l15 = lane & 15;

    // Whh B-fragments -> registers (this wave's 4 gate tiles x 4 K-steps)
    const __bf16* Wd = whh + (size_t)dir * G4 * HID;
    v16bf bfrag[4][4];
    #pragma unroll
    for (int q = 0; q < 4; ++q)
        #pragma unroll
        for (int ks = 0; ks < 4; ++ks)
            bfrag[q][ks] = *(const v16bf*)(Wd + (size_t)(q * HID + wv * 16 + l15) * HID
                                               + ks * 32 + lh * 16);

    for (int i = tid; i < 16 * LDS_PAD; i += 256) hlds[i] = (__bf16)0.0f;

    const float* gsrc = gx + (size_t)dir * T_LEN * (size_t)G4 * NB;

#if USE_ASYNC_GX
    {   // preload gx(t0) into buffer 0: 512 rows x 16 floats, 16B per lane-chunk
        const int t0 = dir ? (T_LEN - 1) : 0;
        #pragma unroll
        for (int k = 0; k < 8; ++k) {
            int c = tid + k * 256;         // chunk 0..2047
            int g = c >> 2, part = c & 3;
            async_copy16(gsrc + ((size_t)t0 * G4 + g) * NB + b0 + part * 4,
                         gxlds + g * 16 + part * 4);
        }
    }
#endif
    __syncthreads();

    v8f cf = {0.f, 0.f, 0.f, 0.f, 0.f, 0.f, 0.f, 0.f};   // cell state, register-resident

    for (int step = 0; step < T_LEN; ++step) {
        const int t   = dir ? (T_LEN - 1 - step) : step;
        const int cur = step & 1;

#if USE_ASYNC_GX
        if (step + 1 < T_LEN) {           // prefetch gx(t+1) into the other buffer
            const int t2 = dir ? (T_LEN - 2 - step) : (step + 1);
            float* buf = gxlds + (cur ^ 1) * (G4 * 16);
            #pragma unroll
            for (int k = 0; k < 8; ++k) {
                int c = tid + k * 256;
                int g = c >> 2, part = c & 3;
                async_copy16(gsrc + ((size_t)t2 * G4 + g) * NB + b0 + part * 4,
                             buf + g * 16 + part * 4);
            }
        }
#endif

        // A fragments for h_{t-1}: K = 128 in 4 chunks of 32
        v16bf a[4];
        #pragma unroll
        for (int ks = 0; ks < 4; ++ks) {
            const __bf16* hp = hlds + l15 * LDS_PAD + ks * 32 + lh * 16;
            v8bf lo = *(const v8bf*)hp;
            v8bf hi = *(const v8bf*)(hp + 8);
            #pragma unroll
            for (int e = 0; e < 8; ++e) { a[ks][e] = lo[e]; a[ks][8 + e] = hi[e]; }
        }

        // init accumulators with precomputed input gates
        v8f acc[4];
#if USE_ASYNC_GX
        if (step + 1 < T_LEN) __builtin_amdgcn_s_wait_asynccnt(8);  // cur buffer landed
        else                  __builtin_amdgcn_s_wait_asynccnt(0);
        __syncthreads();                  // all waves' async copies visible
        {
            const float* gp = gxlds + cur * (G4 * 16);
            #pragma unroll
            for (int q = 0; q < 4; ++q)
                acc[q] = *(const v8f*)(gp + (q * HID + wv * 16 + l15) * 16 + lh * 8);
        }
#else
        #pragma unroll
        for (int q = 0; q < 4; ++q)
            acc[q] = *(const v8f*)(gsrc + ((size_t)t * G4 + q * HID + wv * 16 + l15) * NB
                                        + b0 + lh * 8);
#endif

        // gates += h @ WhhT  (i,f,g,o tiles, B resident in VGPRs)
        #pragma unroll
        for (int q = 0; q < 4; ++q)
            #pragma unroll
            for (int ks = 0; ks < 4; ++ks)
                acc[q] = __builtin_amdgcn_wmma_f32_16x16x32_bf16(
                             false, a[ks], false, bfrag[q][ks], (short)0, acc[q], false, false);

        // elementwise cell update in accumulator-fragment layout
        v8f hv;
        #pragma unroll
        for (int v = 0; v < 8; ++v) {
            float ig = sigf(acc[0][v]);
            float fg = sigf(acc[1][v]);
            float gg = tanhfast(acc[2][v]);
            float og = sigf(acc[3][v]);
            float c  = fg * cf[v] + ig * gg;
            cf[v] = c;
            hv[v] = og * tanhfast(c);
        }

        __syncthreads();   // all waves' hlds reads consumed (by WMMA) before overwrite
        #pragma unroll
        for (int v = 0; v < 8; ++v) {
            const int row = lh * 8 + v;       // batch-local (fragment M)
            const int col = wv * 16 + l15;    // hidden unit  (fragment N)
            __bf16 hb = (__bf16)hv[v];
            hlds[row * LDS_PAD + col] = hb;
            hout[((size_t)(b0 + row) * T_LEN + t) * 256 + dir * HID + col] = hb;
        }
        __syncthreads();
    }
}

// ---------------- temporal mean + linear head ----------------
__global__ __launch_bounds__(256) void final_head(
        const __bf16* __restrict__ h,    // [NB][T][256]
        const float* __restrict__ wlin,  // [256]
        const float* __restrict__ blin,  // [1]
        float* __restrict__ out) {       // [NB]
    const int b = blockIdx.x, c = threadIdx.x;
    const __bf16* p = h + (size_t)b * T_LEN * 256 + c;
    float s = 0.f;
    for (int t = 0; t < T_LEN; ++t) s += (float)p[(size_t)t * 256];
    __shared__ float red[256];
    red[c] = s * (1.0f / T_LEN) * wlin[c];
    __syncthreads();
    for (int off = 128; off > 0; off >>= 1) {
        if (c < off) red[c] += red[c + off];
        __syncthreads();
    }
    if (c == 0) out[b] = red[0] + blin[0];
}

// ---------------- host orchestration ----------------
extern "C" void kernel_launch(void* const* d_in, const int* in_sizes, int n_in,
                              void* d_out, int out_size, void* d_ws, size_t ws_size,
                              hipStream_t stream) {
    (void)in_sizes; (void)n_in; (void)out_size; (void)ws_size;
    const float* x = (const float*)d_in[0];
    const float *Wih[3][2], *Whh[3][2], *bih[3][2], *bhh[3][2];
    for (int l = 0; l < 3; ++l)
        for (int d = 0; d < 2; ++d) {
            int base = 1 + (l * 2 + d) * 4;
            Wih[l][d] = (const float*)d_in[base + 0];
            Whh[l][d] = (const float*)d_in[base + 1];
            bih[l][d] = (const float*)d_in[base + 2];
            bhh[l][d] = (const float*)d_in[base + 3];
        }
    const float* Wlin = (const float*)d_in[25];
    const float* blin = (const float*)d_in[26];

    // workspace carve-up (~1.26 GB)
    char* ws = (char*)d_ws;
    const size_t HBE = (size_t)NB * T_LEN * 256;       // bf16 elems per h buffer
    const size_t GXE = (size_t)2 * T_LEN * G4 * NB;    // f32 elems of gx
    __bf16* hA   = (__bf16*)ws;
    __bf16* hB   = (__bf16*)(ws + 2 * HBE);
    float*  gx   = (float*)(ws + 4 * HBE);
    __bf16* whhb = (__bf16*)(ws + 4 * HBE + 4 * GXE);   // [3][2][512][128]
    __bf16* wihb = whhb + (size_t)6 * G4 * HID;         // [2][2][512][256]
    float*  biasc = (float*)(wihb + (size_t)4 * G4 * 256); // [3][2][512]

    // weight prep (bf16 conversion + fused biases)
    for (int l = 0; l < 3; ++l)
        for (int d = 0; d < 2; ++d) {
            f2bf_kernel<<<(G4 * HID + 255) / 256, 256, 0, stream>>>(
                Whh[l][d], whhb + (size_t)(l * 2 + d) * G4 * HID, G4 * HID);
            bias_kernel<<<(G4 + 255) / 256, 256, 0, stream>>>(
                bih[l][d], bhh[l][d], biasc + (l * 2 + d) * G4, G4);
        }
    for (int l = 1; l < 3; ++l)
        for (int d = 0; d < 2; ++d)
            f2bf_kernel<<<(G4 * 256 + 255) / 256, 256, 0, stream>>>(
                Wih[l][d], wihb + (size_t)((l - 1) * 2 + d) * G4 * 256, G4 * 256);

    // LDS: h tile + double-buffered gx slice (Whh register-resident)
    const int LDSB = ((16 * LDS_PAD * 2 + 15) & ~15) + 2 * G4 * 16 * 4;

    // layer 0
    in_proj_l0<<<(unsigned)((size_t)T_LEN * G4 * NB / 256), 256, 0, stream>>>(
        x, Wih[0][0], Wih[0][1], biasc, gx);
    lstm_scan<<<8, 256, LDSB, stream>>>(gx, whhb, hA);
    // layer 1
    in_proj_gemm<<<dim3(T_LEN * 4, 2), 256, 0, stream>>>(hA, wihb, biasc + 2 * G4, gx);
    lstm_scan<<<8, 256, LDSB, stream>>>(gx, whhb + (size_t)2 * G4 * HID, hB);
    // layer 2
    in_proj_gemm<<<dim3(T_LEN * 4, 2), 256, 0, stream>>>(hB, wihb + (size_t)2 * G4 * 256,
                                                         biasc + 4 * G4, gx);
    lstm_scan<<<8, 256, LDSB, stream>>>(gx, whhb + (size_t)4 * G4 * HID, hA);
    // head
    final_head<<<NB, 256, 0, stream>>>(hA, Wlin, blin, (float*)d_out);
}
